// MyKernel_57226144252422
// MI455X (gfx1250) — matrix operand order repaired and checked by reading, not compiled, not run on hardware
//
#include <hip/hip_runtime.h>
#include <hip/hip_bf16.h>

typedef __attribute__((ext_vector_type(16))) __bf16          v16bf;
typedef __attribute__((ext_vector_type(8)))  float           v8f;
typedef __attribute__((ext_vector_type(8)))  unsigned short  v8us;
typedef __attribute__((ext_vector_type(16))) unsigned short  v16us;

#define TILE 128
#define LSTR 40   // padded LDS row stride in ushorts (80 B: 16B-aligned, conflict-free)

__device__ __forceinline__ unsigned short f2bf(float f) {
    unsigned u = __builtin_bit_cast(unsigned, f);
    u += 0x7FFFu + ((u >> 16) & 1u);          // round-to-nearest-even
    return (unsigned short)(u >> 16);
}
__device__ __forceinline__ float bf2f(unsigned short h) {
    return __builtin_bit_cast(float, ((unsigned)h) << 16);
}

// Assemble a 16-element bf16 fragment from two aligned 16-byte LDS runs.
__device__ __forceinline__ v16bf lds_frag(const unsigned short* p0,
                                          const unsigned short* p1) {
    v8us a = *(const v8us*)p0;
    v8us b = *(const v8us*)p1;
    v16us c = __builtin_shufflevector(a, b, 0,1,2,3,4,5,6,7,8,9,10,11,12,13,14,15);
    return __builtin_bit_cast(v16bf, c);
}

// CDNA5 async global->LDS copy (16 B per lane), ASYNCcnt-tracked.
__device__ __forceinline__ void async_b128(const void* g, void* l) {
    unsigned lds32 = (unsigned)(uintptr_t)l;              // low 32 bits = LDS offset
    unsigned long long ga = (unsigned long long)(uintptr_t)g;
    asm volatile("global_load_async_to_lds_b128 %0, %1, off"
                 :: "v"(lds32), "v"(ga) : "memory");
}
#define S_WAIT_ASYNC_8() asm volatile("s_wait_asynccnt 0x8" ::: "memory")
#define S_WAIT_ASYNC_0() asm volatile("s_wait_asynccnt 0x0" ::: "memory")

// ---------------------------------------------------------------------------
// Kernel: exact fp32 row norms of x (N x D) and y (M x D) into workspace.
// ---------------------------------------------------------------------------
__global__ void row_norms_kernel(const float* __restrict__ x,
                                 const float* __restrict__ y,
                                 float* __restrict__ a2, float* __restrict__ b2,
                                 int N, int M, int D) {
    int i = blockIdx.x * blockDim.x + threadIdx.x;
    if (i < N) {
        const float4* p = (const float4*)(x + (size_t)i * D);
        float s = 0.f;
        for (int j = 0; j < (D >> 2); ++j) {
            float4 v = p[j];
            s += v.x*v.x + v.y*v.y + v.z*v.z + v.w*v.w;
        }
        a2[i] = s;
    }
    if (i < M) {
        const float4* p = (const float4*)(y + (size_t)i * D);
        float s = 0.f;
        for (int j = 0; j < (D >> 2); ++j) {
            float4 v = p[j];
            s += v.x*v.x + v.y*v.y + v.z*v.z + v.w*v.w;
        }
        b2[i] = s;
    }
}

// ---------------------------------------------------------------------------
// Kernel: one-time fp32 -> (hi,lo) bf16 split pack (8 elements per thread).
// ---------------------------------------------------------------------------
__global__ void pack_bf16_kernel(const float* __restrict__ src,
                                 unsigned short* __restrict__ hi,
                                 unsigned short* __restrict__ lo,
                                 long long n8) {
    long long i = (long long)blockIdx.x * blockDim.x + threadIdx.x;
    if (i >= n8) return;
    const float4* p = (const float4*)src + i * 2;
    float4 v0 = p[0], v1 = p[1];
    float f[8] = {v0.x, v0.y, v0.z, v0.w, v1.x, v1.y, v1.z, v1.w};
    v8us h, l;
    #pragma unroll
    for (int k = 0; k < 8; ++k) {
        unsigned short hh = f2bf(f[k]);
        h[k] = hh;
        l[k] = f2bf(f[k] - bf2f(hh));
    }
    *(v8us*)(hi + i * 8) = h;
    *(v8us*)(lo + i * 8) = l;
}

// ---------------------------------------------------------------------------
// Fast path: fused dual-RBF with pre-packed bf16 hi/lo planes, async double-
// buffered global->LDS staging, bf16x3-split WMMA GEMM for the D=256 term.
// ---------------------------------------------------------------------------
__global__ __launch_bounds__(256)
void rbf2_wmma_packed(const float* __restrict__ xp, const float* __restrict__ yp,
                      const unsigned short* __restrict__ xhi, const unsigned short* __restrict__ xlo,
                      const unsigned short* __restrict__ yhi, const unsigned short* __restrict__ ylo,
                      const float* __restrict__ sa, const float* __restrict__ sb,
                      const float* __restrict__ a2, const float* __restrict__ b2,
                      float* __restrict__ out, int N, int M, int D) {
    __shared__ unsigned short AhiS[2][TILE * LSTR], AloS[2][TILE * LSTR];
    __shared__ unsigned short BhiS[2][TILE * LSTR], BloS[2][TILE * LSTR];
    __shared__ float sx2[TILE], sy2[TILE];
    __shared__ float sxp[TILE * 2], syp[TILE * 2];

    const int tid  = threadIdx.x;
    const int lane = tid & 31;
    const int w    = tid >> 5;       // wave 0..7
    const int wr   = w >> 1;         // wave row 0..3  (32 rows each)
    const int wc   = w & 1;          // wave col 0..1  (64 cols each)
    const int ln   = lane & 15;
    const int g    = lane >> 4;      // half-wave group

    const int rowBase = blockIdx.y * TILE;
    const int colBase = blockIdx.x * TILE;

    if (tid < 2 * TILE) {
        sxp[tid] = xp[(size_t)rowBase * 2 + tid];
        syp[tid] = yp[(size_t)colBase * 2 + tid];
    }
    if (tid < TILE) {
        sx2[tid] = a2[rowBase + tid];
        sy2[tid] = b2[colBase + tid];
    }

    // Per-thread staging slots: 2 chunks of 8 ushorts per plane.
    const int e0 = tid, e1 = tid + 256;
    const int r0 = e0 >> 2, q0 = e0 & 3;
    const int r1 = e1 >> 2, q1 = e1 & 3;
    const int l0 = r0 * LSTR + q0 * 8;
    const int l1 = r1 * LSTR + q1 * 8;
    const size_t gxr0 = (size_t)(rowBase + r0) * D + q0 * 8;
    const size_t gxr1 = (size_t)(rowBase + r1) * D + q1 * 8;
    const size_t gyr0 = (size_t)(colBase + r0) * D + q0 * 8;
    const size_t gyr1 = (size_t)(colBase + r1) * D + q1 * 8;

    #define ISSUE_STAGE(kb, b)                                            \
        do {                                                              \
            async_b128(xhi + gxr0 + (kb), &AhiS[b][l0]);                  \
            async_b128(xlo + gxr0 + (kb), &AloS[b][l0]);                  \
            async_b128(yhi + gyr0 + (kb), &BhiS[b][l0]);                  \
            async_b128(ylo + gyr0 + (kb), &BloS[b][l0]);                  \
            async_b128(xhi + gxr1 + (kb), &AhiS[b][l1]);                  \
            async_b128(xlo + gxr1 + (kb), &AloS[b][l1]);                  \
            async_b128(yhi + gyr1 + (kb), &BhiS[b][l1]);                  \
            async_b128(ylo + gyr1 + (kb), &BloS[b][l1]);                  \
        } while (0)

    v8f acc[2][4];
    #pragma unroll
    for (int i = 0; i < 2; ++i)
        #pragma unroll
        for (int j = 0; j < 4; ++j)
            acc[i][j] = v8f{0.f,0.f,0.f,0.f,0.f,0.f,0.f,0.f};

    const int nk = D >> 5;           // K-steps of 32
    ISSUE_STAGE(0, 0);

    for (int it = 0; it < nk; ++it) {
        const int cur = it & 1;
        if (it + 1 < nk) {
            ISSUE_STAGE((it + 1) << 5, 1 - cur);
            S_WAIT_ASYNC_8();        // current stage's 8 transfers complete
        } else {
            S_WAIT_ASYNC_0();
        }
        __syncthreads();             // all waves' stage data visible

        // A (16x32 bf16): j<8 -> K=g*8+j ; j>=8 -> K=16+g*8+(j-8)
        // B (32x16 bf16): j<8 -> K=g*8+j ; j>=8 -> K=16+g*8+(j-8)
        v16bf aH[2], aL[2], bH[4], bL[4];
        #pragma unroll
        for (int ti = 0; ti < 2; ++ti) {
            int m  = wr * 32 + ti * 16 + ln;
            int o0 = m * LSTR + g * 8;
            aH[ti] = lds_frag(&AhiS[cur][o0], &AhiS[cur][o0 + 16]);
            aL[ti] = lds_frag(&AloS[cur][o0], &AloS[cur][o0 + 16]);
        }
        #pragma unroll
        for (int tj = 0; tj < 4; ++tj) {
            int n  = wc * 64 + tj * 16 + ln;
            int o0 = n * LSTR + g * 8;
            bH[tj] = lds_frag(&BhiS[cur][o0], &BhiS[cur][o0 + 16]);
            bL[tj] = lds_frag(&BloS[cur][o0], &BloS[cur][o0 + 16]);
        }

        #pragma unroll
        for (int ti = 0; ti < 2; ++ti)
            #pragma unroll
            for (int tj = 0; tj < 4; ++tj)
                acc[ti][tj] = __builtin_amdgcn_wmma_f32_16x16x32_bf16(
                    false, aH[ti], false, bH[tj], (short)0, acc[ti][tj], false, false);
        #pragma unroll
        for (int ti = 0; ti < 2; ++ti)
            #pragma unroll
            for (int tj = 0; tj < 4; ++tj)
                acc[ti][tj] = __builtin_amdgcn_wmma_f32_16x16x32_bf16(
                    false, aH[ti], false, bL[tj], (short)0, acc[ti][tj], false, false);
        #pragma unroll
        for (int ti = 0; ti < 2; ++ti)
            #pragma unroll
            for (int tj = 0; tj < 4; ++tj)
                acc[ti][tj] = __builtin_amdgcn_wmma_f32_16x16x32_bf16(
                    false, aL[ti], false, bH[tj], (short)0, acc[ti][tj], false, false);

        __syncthreads();             // all reads of buf[cur] done before refill
    }
    #undef ISSUE_STAGE

    const float va_s = sa[0];
    const float vb_s = sb[0];
    const float ga = 0.5f / (va_s * va_s);
    const float gb = 0.5f / (vb_s * vb_s);

    // C/D layout -> element r of v8f: M = r + 8*g, N = ln.
    #pragma unroll
    for (int ti = 0; ti < 2; ++ti) {
        #pragma unroll
        for (int tj = 0; tj < 4; ++tj) {
            int nl = wc * 64 + tj * 16 + ln;
            float qy0 = syp[2 * nl], qy1 = syp[2 * nl + 1];
            float q2  = qy0 * qy0 + qy1 * qy1;
            float y2  = sy2[nl];
            #pragma unroll
            for (int r = 0; r < 8; ++r) {
                int ml = wr * 32 + ti * 16 + r + 8 * g;
                float px0 = sxp[2 * ml], px1 = sxp[2 * ml + 1];
                float dp  = px0 * qy0 + px1 * qy1;
                float d2a = fmaxf(px0 * px0 + px1 * px1 + q2 - 2.f * dp, 0.f);
                float d2b = fmaxf(sx2[ml] + y2 - 2.f * acc[ti][tj][r], 0.f);
                out[(size_t)(rowBase + ml) * M + (colBase + nl)] =
                    __expf(-ga * d2a) + __expf(-gb * d2b);
            }
        }
    }
}

// ---------------------------------------------------------------------------
// Fallback (small workspace): round-1 kernel with in-kernel conversion.
// ---------------------------------------------------------------------------
__global__ __launch_bounds__(256)
void rbf2_wmma_kernel(const float* __restrict__ xp, const float* __restrict__ yp,
                      const float* __restrict__ x,  const float* __restrict__ y,
                      const float* __restrict__ sa, const float* __restrict__ sb,
                      const float* __restrict__ a2, const float* __restrict__ b2,
                      float* __restrict__ out, int N, int M, int D) {
    __shared__ unsigned short Ahi[TILE * LSTR], Alo[TILE * LSTR];
    __shared__ unsigned short Bhi[TILE * LSTR], Blo[TILE * LSTR];
    __shared__ float sx2[TILE], sy2[TILE];
    __shared__ float sxp[TILE * 2], syp[TILE * 2];

    const int tid  = threadIdx.x;
    const int lane = tid & 31;
    const int w    = tid >> 5;
    const int wr   = w >> 1;
    const int wc   = w & 1;
    const int ln   = lane & 15;
    const int g    = lane >> 4;

    const int rowBase = blockIdx.y * TILE;
    const int colBase = blockIdx.x * TILE;

    if (tid < 2 * TILE) {
        sxp[tid] = xp[(size_t)rowBase * 2 + tid];
        syp[tid] = yp[(size_t)colBase * 2 + tid];
    }
    if (tid < TILE) {
        sx2[tid] = a2[rowBase + tid];
        sy2[tid] = b2[colBase + tid];
    }

    v8f acc[2][4];
    #pragma unroll
    for (int i = 0; i < 2; ++i)
        #pragma unroll
        for (int j = 0; j < 4; ++j)
            acc[i][j] = v8f{0.f,0.f,0.f,0.f,0.f,0.f,0.f,0.f};

    for (int kb = 0; kb < D; kb += 32) {
        __syncthreads();
        #pragma unroll
        for (int i = 0; i < 4; ++i) {
            int e  = tid + i * 256;
            int r  = e >> 3, c4 = e & 7;
            const float4 va = *(const float4*)(x + (size_t)(rowBase + r) * D + kb + c4 * 4);
            const float4 vb = *(const float4*)(y + (size_t)(colBase + r) * D + kb + c4 * 4);
            int o = r * LSTR + c4 * 4;
            float fa[4] = {va.x, va.y, va.z, va.w};
            float fb[4] = {vb.x, vb.y, vb.z, vb.w};
            #pragma unroll
            for (int k = 0; k < 4; ++k) {
                unsigned short h = f2bf(fa[k]);
                Ahi[o + k] = h;
                Alo[o + k] = f2bf(fa[k] - bf2f(h));
                h = f2bf(fb[k]);
                Bhi[o + k] = h;
                Blo[o + k] = f2bf(fb[k] - bf2f(h));
            }
        }
        __syncthreads();

        v16bf aH[2], aL[2], bH[4], bL[4];
        #pragma unroll
        for (int ti = 0; ti < 2; ++ti) {
            int m  = wr * 32 + ti * 16 + ln;
            int o0 = m * LSTR + g * 8;
            aH[ti] = lds_frag(&Ahi[o0], &Ahi[o0 + 16]);
            aL[ti] = lds_frag(&Alo[o0], &Alo[o0 + 16]);
        }
        #pragma unroll
        for (int tj = 0; tj < 4; ++tj) {
            int n  = wc * 64 + tj * 16 + ln;
            int o0 = n * LSTR + g * 8;
            bH[tj] = lds_frag(&Bhi[o0], &Bhi[o0 + 16]);
            bL[tj] = lds_frag(&Blo[o0], &Blo[o0 + 16]);
        }

        #pragma unroll
        for (int ti = 0; ti < 2; ++ti)
            #pragma unroll
            for (int tj = 0; tj < 4; ++tj)
                acc[ti][tj] = __builtin_amdgcn_wmma_f32_16x16x32_bf16(
                    false, aH[ti], false, bH[tj], (short)0, acc[ti][tj], false, false);
        #pragma unroll
        for (int ti = 0; ti < 2; ++ti)
            #pragma unroll
            for (int tj = 0; tj < 4; ++tj)
                acc[ti][tj] = __builtin_amdgcn_wmma_f32_16x16x32_bf16(
                    false, aH[ti], false, bL[tj], (short)0, acc[ti][tj], false, false);
        #pragma unroll
        for (int ti = 0; ti < 2; ++ti)
            #pragma unroll
            for (int tj = 0; tj < 4; ++tj)
                acc[ti][tj] = __builtin_amdgcn_wmma_f32_16x16x32_bf16(
                    false, aL[ti], false, bH[tj], (short)0, acc[ti][tj], false, false);
    }

    const float va_s = sa[0];
    const float vb_s = sb[0];
    const float ga = 0.5f / (va_s * va_s);
    const float gb = 0.5f / (vb_s * vb_s);

    #pragma unroll
    for (int ti = 0; ti < 2; ++ti) {
        #pragma unroll
        for (int tj = 0; tj < 4; ++tj) {
            int nl = wc * 64 + tj * 16 + ln;
            float qy0 = syp[2 * nl], qy1 = syp[2 * nl + 1];
            float q2  = qy0 * qy0 + qy1 * qy1;
            float y2  = sy2[nl];
            #pragma unroll
            for (int r = 0; r < 8; ++r) {
                int ml = wr * 32 + ti * 16 + r + 8 * g;
                float px0 = sxp[2 * ml], px1 = sxp[2 * ml + 1];
                float dp  = px0 * qy0 + px1 * qy1;
                float d2a = fmaxf(px0 * px0 + px1 * px1 + q2 - 2.f * dp, 0.f);
                float d2b = fmaxf(sx2[ml] + y2 - 2.f * acc[ti][tj][r], 0.f);
                out[(size_t)(rowBase + ml) * M + (colBase + nl)] =
                    __expf(-ga * d2a) + __expf(-gb * d2b);
            }
        }
    }
}

extern "C" void kernel_launch(void* const* d_in, const int* in_sizes, int n_in,
                              void* d_out, int out_size, void* d_ws, size_t ws_size,
                              hipStream_t stream) {
    const float* xp = (const float*)d_in[0];
    const float* yp = (const float*)d_in[1];
    const float* x  = (const float*)d_in[2];
    const float* y  = (const float*)d_in[3];
    const float* sa = (const float*)d_in[4];
    const float* sb = (const float*)d_in[5];
    float* out = (float*)d_out;

    const int N = in_sizes[0] / 2;          // x_proj is [N, 2]
    const int M = in_sizes[1] / 2;          // y_proj is [M, 2]
    const int D = in_sizes[2] / N;          // x is [N, D]

    float* a2 = (float*)d_ws;               // N floats
    float* b2 = a2 + N;                     // M floats

    const int threads = 256;
    const int mx = (N > M) ? N : M;
    row_norms_kernel<<<(mx + threads - 1) / threads, threads, 0, stream>>>(
        x, y, a2, b2, N, M, D);

    const size_t normBytes = (size_t)(N + M) * sizeof(float);
    const size_t packBytes = ((size_t)N * D + (size_t)M * D) * 2 * sizeof(unsigned short);
    dim3 grid(M / TILE, N / TILE);

    if (ws_size >= normBytes + packBytes) {
        unsigned short* xhi = (unsigned short*)(b2 + M);
        unsigned short* xlo = xhi + (size_t)N * D;
        unsigned short* yhi = xlo + (size_t)N * D;
        unsigned short* ylo = yhi + (size_t)M * D;

        long long n8x = (long long)N * D / 8;
        long long n8y = (long long)M * D / 8;
        pack_bf16_kernel<<<(unsigned)((n8x + threads - 1) / threads), threads, 0, stream>>>(
            x, xhi, xlo, n8x);
        pack_bf16_kernel<<<(unsigned)((n8y + threads - 1) / threads), threads, 0, stream>>>(
            y, yhi, ylo, n8y);

        rbf2_wmma_packed<<<grid, threads, 0, stream>>>(
            xp, yp, xhi, xlo, yhi, ylo, sa, sb, a2, b2, out, N, M, D);
    } else {
        rbf2_wmma_kernel<<<grid, threads, 0, stream>>>(
            xp, yp, x, y, sa, sb, a2, b2, out, N, M, D);
    }
}
